// MultiHeadSelfAttention2D_1468878815931
// MI455X (gfx1250) — matrix operand order repaired and checked by reading, not compiled, not run on hardware
//
#include <hip/hip_runtime.h>
#include <math.h>

// ---------------------------------------------------------------------------
// MultiHeadSelfAttention2D for MI455X (gfx1250, wave32, WMMA)
// Flash-attention formulation, f16 WMMA (v_wmma_f32_16x16x32_f16) for both
// QK^T and P*V; K/V streamed from the 192MB L2 (whole K/V set ~64MB).
// ---------------------------------------------------------------------------

typedef __attribute__((ext_vector_type(16))) _Float16 v16h;
typedef __attribute__((ext_vector_type(8)))  _Float16 v8h;
typedef __attribute__((ext_vector_type(8)))  float    v8f;

#define B_   2
#define C_   64
#define T_   3000
#define F_   65
#define H_   4
#define D_   4
#define E_   16
#define DF_  260           // D*F (q/k embedding)
#define EF_  1040          // E*F (v embedding) == 65 tiles of 16
#define DP_  288           // DF_ padded to multiple of 32 (9 K-chunks)
#define TP_  3008          // T padded to multiple of 32
#define NPOS (B_*T_*F_)    // 390000 positions
#define NT_  65            // EF_/16 N-tiles
#define SCALE_ 0.06201736729460423f   // 1/sqrt(260)
#define EPS_  1e-5f
#define NEG_  -3.0e38f

// ---- fragment loaders (16-bit WMMA operands; A and B must share one k order) ---------------
// A (16x32 f16): lane L: M=L&15; halves [k0..k0+7] and [k0+16..k0+23], k0=(L>>4)*8
__device__ __forceinline__ v16h load_a16(const _Float16* p) {
  v8h lo = *(const v8h*)(p);
  v8h hi = *(const v8h*)(p + 16);
  v16h r;
#pragma unroll
  for (int i = 0; i < 8; ++i) { r[i] = lo[i]; r[i + 8] = hi[i]; }
  return r;
}
// contiguous k order, this file's choice for both A and B: lane L: row/col=L&15; 16 halves at K=(L>>4)*16
__device__ __forceinline__ v16h load_b16(const _Float16* p) {
  v8h lo = *(const v8h*)(p);
  v8h hi = *(const v8h*)(p + 8);
  v16h r;
#pragma unroll
  for (int i = 0; i < 8; ++i) { r[i] = lo[i]; r[i + 8] = hi[i]; }
  return r;
}

// ---------------------------------------------------------------------------
// Kernel 0: zero the pad regions (K/Q pad rows, Vt pad columns) so no
// NaN/garbage can enter the WMMA pipes on the first launch.
// total elements per bh: 2*(8*288) + 1040*8 = 12928 ; 8 bh slices.
// ---------------------------------------------------------------------------
__global__ __launch_bounds__(256) void pad_zero_kernel(_Float16* __restrict__ Qf,
                                                       _Float16* __restrict__ Kf,
                                                       _Float16* __restrict__ Vt) {
  int idx = blockIdx.x * 256 + threadIdx.x;
  int bh = idx / 12928;
  if (bh >= B_ * H_) return;
  int r = idx - bh * 12928;
  if (r < 2304) {                      // Qf pad rows t in [T_, TP_)
    int row = r / DP_, col = r - row * DP_;
    Qf[((size_t)bh * TP_ + T_ + row) * DP_ + col] = (_Float16)0.f;
  } else if (r < 4608) {               // Kf pad rows
    r -= 2304;
    int row = r / DP_, col = r - row * DP_;
    Kf[((size_t)bh * TP_ + T_ + row) * DP_ + col] = (_Float16)0.f;
  } else {                             // Vt pad columns t in [T_, TP_)
    r -= 4608;
    int row = r >> 3, c = r & 7;
    Vt[((size_t)bh * EF_ + row) * TP_ + T_ + c] = (_Float16)0.f;
  }
}

// ---------------------------------------------------------------------------
// Kernel 1: fused Q/K/V 1x1-conv + PReLU + per-head channel LayerNorm.
// One thread per (b,t,f). Writes f16:
//   Qf,Kf : [bh][Tp][288]  (row-major, cols 260..287 zeroed)
//   Vt    : [bh][1040][Tp] (transposed so attention B-fragments are contiguous)
// ---------------------------------------------------------------------------
__global__ __launch_bounds__(256) void proj_kernel(
    const float* __restrict__ x,
    const float* __restrict__ Wq, const float* __restrict__ bq, const float* __restrict__ aq,
    const float* __restrict__ gq, const float* __restrict__ btq,
    const float* __restrict__ Wk, const float* __restrict__ bk, const float* __restrict__ ak,
    const float* __restrict__ gk, const float* __restrict__ btk,
    const float* __restrict__ Wv, const float* __restrict__ bv, const float* __restrict__ av,
    const float* __restrict__ gv, const float* __restrict__ btv,
    _Float16* __restrict__ Qf, _Float16* __restrict__ Kf, _Float16* __restrict__ Vt) {
  __shared__ float sWq[16 * 64], sWk[16 * 64], sWv[64 * 64];
  __shared__ float sbq[16], sgq[16], sbtq[16];
  __shared__ float sbk[16], sgk[16], sbtk[16];
  __shared__ float sbv[64], sgv[64], sbtv[64];
  __shared__ float saq[4], sak[4], sav[4];
  const int tid = threadIdx.x;
  for (int i = tid; i < 1024; i += 256) { sWq[i] = Wq[i]; sWk[i] = Wk[i]; }
  for (int i = tid; i < 4096; i += 256) { sWv[i] = Wv[i]; }
  if (tid < 16) {
    sbq[tid] = bq[tid]; sgq[tid] = gq[tid]; sbtq[tid] = btq[tid];
    sbk[tid] = bk[tid]; sgk[tid] = gk[tid]; sbtk[tid] = btk[tid];
  }
  if (tid < 64) { sbv[tid] = bv[tid]; sgv[tid] = gv[tid]; sbtv[tid] = btv[tid]; }
  if (tid < 4)  { saq[tid] = aq[tid]; sak[tid] = ak[tid]; sav[tid] = av[tid]; }
  __syncthreads();

  int p = blockIdx.x * 256 + tid;
  if (p >= NPOS) return;
  int b = p / (T_ * F_);
  int r = p - b * (T_ * F_);
  int t = r / F_;
  int f = r - t * F_;

  float qa[16], ka[16], va[64];
#pragma unroll
  for (int j = 0; j < 16; ++j) { qa[j] = 0.f; ka[j] = 0.f; }
#pragma unroll
  for (int j = 0; j < 64; ++j) va[j] = 0.f;

  const float* xp = x + ((size_t)b * C_) * T_ * F_ + (size_t)t * F_ + f;
#pragma unroll 4
  for (int c = 0; c < C_; ++c) {
    float xc = xp[(size_t)c * T_ * F_];
#pragma unroll
    for (int j = 0; j < 16; ++j) { qa[j] += sWq[j * 64 + c] * xc; ka[j] += sWk[j * 64 + c] * xc; }
#pragma unroll
    for (int j = 0; j < 64; ++j) { va[j] += sWv[j * 64 + c] * xc; }
  }

#pragma unroll
  for (int h = 0; h < H_; ++h) {
    const int bh = b * H_ + h;
    _Float16* qrow = Qf + ((size_t)bh * TP_ + t) * DP_;
    _Float16* krow = Kf + ((size_t)bh * TP_ + t) * DP_;
    // ---- Q: bias, PReLU, LN over D=4 ----
    {
      float v[4], mu = 0.f;
#pragma unroll
      for (int d = 0; d < 4; ++d) {
        float y = qa[h * 4 + d] + sbq[h * 4 + d];
        y = (y >= 0.f) ? y : saq[h] * y;
        v[d] = y; mu += y;
      }
      mu *= 0.25f;
      float var = 0.f;
#pragma unroll
      for (int d = 0; d < 4; ++d) { float dd = v[d] - mu; var += dd * dd; }
      float inv = rsqrtf(var * 0.25f + EPS_);
#pragma unroll
      for (int d = 0; d < 4; ++d)
        qrow[d * F_ + f] = (_Float16)((v[d] - mu) * inv * sgq[h * 4 + d] + sbtq[h * 4 + d]);
    }
    // ---- K ----
    {
      float v[4], mu = 0.f;
#pragma unroll
      for (int d = 0; d < 4; ++d) {
        float y = ka[h * 4 + d] + sbk[h * 4 + d];
        y = (y >= 0.f) ? y : sak[h] * y;
        v[d] = y; mu += y;
      }
      mu *= 0.25f;
      float var = 0.f;
#pragma unroll
      for (int d = 0; d < 4; ++d) { float dd = v[d] - mu; var += dd * dd; }
      float inv = rsqrtf(var * 0.25f + EPS_);
#pragma unroll
      for (int d = 0; d < 4; ++d)
        krow[d * F_ + f] = (_Float16)((v[d] - mu) * inv * sgk[h * 4 + d] + sbtk[h * 4 + d]);
    }
    // ---- V: LN over E=16, write transposed ----
    {
      float v[16], mu = 0.f;
#pragma unroll
      for (int e = 0; e < 16; ++e) {
        float y = va[h * 16 + e] + sbv[h * 16 + e];
        y = (y >= 0.f) ? y : sav[h] * y;
        v[e] = y; mu += y;
      }
      mu *= (1.f / 16.f);
      float var = 0.f;
#pragma unroll
      for (int e = 0; e < 16; ++e) { float dd = v[e] - mu; var += dd * dd; }
      float inv = rsqrtf(var * (1.f / 16.f) + EPS_);
#pragma unroll
      for (int e = 0; e < 16; ++e)
        Vt[((size_t)bh * EF_ + (e * F_ + f)) * TP_ + t] =
            (_Float16)((v[e] - mu) * inv * sgv[h * 16 + e] + sbtv[h * 16 + e]);
    }
    // zero the d-padding columns 260..287 of this row
    if (f < DP_ - DF_) { qrow[DF_ + f] = (_Float16)0.f; krow[DF_ + f] = (_Float16)0.f; }
  }
}

// ---------------------------------------------------------------------------
// Kernel 2: flash attention per (bh, 16-query tile). 8 waves / 256 threads.
// Wave 0: S = Q K^T (two 16x16 wmma chains over 9 K-chunks) + online softmax,
//         publishes P (f16) and row rescale factors via LDS.
// Waves 1..7: own ~10 of the 65 O N-tiles each; rescale + wmma-accumulate P*V.
// ---------------------------------------------------------------------------
__global__ __launch_bounds__(256) void attn_kernel(const _Float16* __restrict__ Qf,
                                                   const _Float16* __restrict__ Kf,
                                                   const _Float16* __restrict__ Vt,
                                                   _Float16* __restrict__ Of) {
  __shared__ __align__(16) _Float16 Plds[16][32];
  __shared__ float alphaLds[16];
  __shared__ float lLds[16];

  const int tid = threadIdx.x;
  const int warp = tid >> 5;
  const int lane = tid & 31;
  const int l15 = lane & 15;
  const int hi16 = lane >> 4;          // 0: rows 0-7 / K-low ; 1: rows 8-15 / K-high
  const int bh = blockIdx.y;
  const int t0 = blockIdx.x * 16;

  v16h qfrag[9];                       // wave-0 resident Q A-fragments
  float mrun[8], lrun[8];              // wave-0 per-row running max / sum
  v8f  oacc[10];                       // waves 1..7 O accumulators (C-layout)

  if (warp == 0) {
    const _Float16* qrow = Qf + ((size_t)bh * TP_ + (t0 + l15)) * DP_;
#pragma unroll
    for (int c = 0; c < 9; ++c) qfrag[c] = load_b16(qrow + c * 32 + hi16 * 16);
#pragma unroll
    for (int r = 0; r < 8; ++r) { mrun[r] = NEG_; lrun[r] = 0.f; }
  } else {
#pragma unroll
    for (int j = 0; j < 10; ++j) {
#pragma unroll
      for (int r = 0; r < 8; ++r) oacc[j][r] = 0.f;
    }
  }

  for (int s0 = 0; s0 < TP_; s0 += 32) {
    if (warp == 0) {
      // ---- S = Q K^T for 32 keys (two 16-col fragments) ----
      v8f sa = {}, sb = {};
      const _Float16* krow0 = Kf + ((size_t)bh * TP_ + (s0 + l15)) * DP_;
      const _Float16* krow1 = krow0 + (size_t)16 * DP_;
      if (s0 + 32 < TP_)  // hint next key block toward L0/L2
        __builtin_prefetch(krow0 + (size_t)32 * DP_, 0, 1);
#pragma unroll
      for (int c = 0; c < 9; ++c) {
        v16h b0 = load_b16(krow0 + c * 32 + hi16 * 16);
        v16h b1 = load_b16(krow1 + c * 32 + hi16 * 16);
        sa = __builtin_amdgcn_wmma_f32_16x16x32_f16(false, qfrag[c], false, b0,
                                                    (short)0, sa, false, false);
        sb = __builtin_amdgcn_wmma_f32_16x16x32_f16(false, qfrag[c], false, b1,
                                                    (short)0, sb, false, false);
      }
      // ---- online softmax (C layout: lane holds one column, 8 rows) ----
      const bool v0 = (s0 + l15) < T_;
      const bool v1 = (s0 + 16 + l15) < T_;
      const int prow = hi16 * 8;
#pragma unroll
      for (int r = 0; r < 8; ++r) {
        float a = v0 ? sa[r] : NEG_;
        float b = v1 ? sb[r] : NEG_;
        float cm = fmaxf(a, b);
        cm = fmaxf(cm, __shfl_xor(cm, 1, 32));
        cm = fmaxf(cm, __shfl_xor(cm, 2, 32));
        cm = fmaxf(cm, __shfl_xor(cm, 4, 32));
        cm = fmaxf(cm, __shfl_xor(cm, 8, 32));
        float mnew = fmaxf(mrun[r], cm);
        float alpha = __expf((mrun[r] - mnew) * SCALE_);
        float p0 = __expf((a - mnew) * SCALE_);
        float p1 = __expf((b - mnew) * SCALE_);
        float ps = p0 + p1;
        ps += __shfl_xor(ps, 1, 32);
        ps += __shfl_xor(ps, 2, 32);
        ps += __shfl_xor(ps, 4, 32);
        ps += __shfl_xor(ps, 8, 32);
        lrun[r] = lrun[r] * alpha + ps;
        mrun[r] = mnew;
        Plds[prow + r][l15]      = (_Float16)p0;
        Plds[prow + r][16 + l15] = (_Float16)p1;
        if (l15 == 0) alphaLds[prow + r] = alpha;
      }
    }
    __syncthreads();
    if (warp != 0) {
      const int w1 = warp - 1;
      float alr[8];
#pragma unroll
      for (int r = 0; r < 8; ++r) alr[r] = alphaLds[hi16 * 8 + r];
      v16h pa = load_b16(&Plds[l15][hi16 * 16]);
#pragma unroll
      for (int j = 0; j < 10; ++j) {
        const int n = w1 + 7 * j;                 // wave-uniform
        if (n < NT_) {
#pragma unroll
          for (int r = 0; r < 8; ++r) oacc[j][r] *= alr[r];
          const _Float16* vrow =
              Vt + ((size_t)bh * EF_ + (n * 16 + l15)) * TP_ + s0 + hi16 * 16;
          v16h vb = load_b16(vrow);
          oacc[j] = __builtin_amdgcn_wmma_f32_16x16x32_f16(false, pa, false, vb,
                                                           (short)0, oacc[j], false, false);
        }
      }
    }
    __syncthreads();
  }

  // ---- epilogue: O /= l ; store f16 ----
  if (warp == 0 && l15 == 0) {
#pragma unroll
    for (int r = 0; r < 8; ++r) lLds[hi16 * 8 + r] = lrun[r];
  }
  __syncthreads();
  if (warp != 0) {
    const int w1 = warp - 1;
    float lrec[8];
#pragma unroll
    for (int r = 0; r < 8; ++r) lrec[r] = 1.f / lLds[hi16 * 8 + r];
#pragma unroll
    for (int j = 0; j < 10; ++j) {
      const int n = w1 + 7 * j;
      if (n < NT_) {
        const int col = n * 16 + l15;
#pragma unroll
        for (int r = 0; r < 8; ++r) {
          const int trow = t0 + hi16 * 8 + r;
          if (trow < T_)
            Of[((size_t)bh * T_ + trow) * EF_ + col] = (_Float16)(oacc[j][r] * lrec[r]);
        }
      }
    }
  }
}

// ---------------------------------------------------------------------------
// Kernel 3: final 64x64 1x1 conv + PReLU + channel LN + residual. One thread
// per (b,t,f); Wp broadcast from LDS; output f32 [B,C,T,F].
// ---------------------------------------------------------------------------
__global__ __launch_bounds__(256) void out_kernel(const float* __restrict__ x,
                                                  const _Float16* __restrict__ Of,
                                                  const float* __restrict__ Wp,
                                                  const float* __restrict__ bp,
                                                  const float* __restrict__ ap,
                                                  const float* __restrict__ gp,
                                                  const float* __restrict__ btp,
                                                  float* __restrict__ out) {
  __shared__ float sWp[64 * 64], sbp[64], sgp[64], sbt[64];
  __shared__ float sap;
  const int tid = threadIdx.x;
  for (int i = tid; i < 4096; i += 256) sWp[i] = Wp[i];
  if (tid < 64) { sbp[tid] = bp[tid]; sgp[tid] = gp[tid]; sbt[tid] = btp[tid]; }
  if (tid == 0) sap = ap[0];
  __syncthreads();

  int p = blockIdx.x * 256 + tid;
  if (p >= NPOS) return;
  int b = p / (T_ * F_);
  int r = p - b * (T_ * F_);
  int t = r / F_;
  int f = r - t * F_;

  float y[64];
#pragma unroll
  for (int o = 0; o < 64; ++o) y[o] = sbp[o];

  const _Float16* obase = Of + ((size_t)(b * H_) * T_ + t) * EF_ + f;
#pragma unroll 2
  for (int c = 0; c < 64; ++c) {
    const int h = c >> 4, e = c & 15;
    float v = (float)obase[(size_t)h * T_ * EF_ + e * F_];
#pragma unroll
    for (int o = 0; o < 64; ++o) y[o] += sWp[o * 64 + c] * v;
  }

  float mu = 0.f;
#pragma unroll
  for (int o = 0; o < 64; ++o) {
    float v = y[o];
    v = (v >= 0.f) ? v : sap * v;
    y[o] = v; mu += v;
  }
  mu *= (1.f / 64.f);
  float var = 0.f;
#pragma unroll
  for (int o = 0; o < 64; ++o) { float d = y[o] - mu; var += d * d; }
  const float inv = rsqrtf(var * (1.f / 64.f) + EPS_);

  const size_t base = ((size_t)b * C_) * T_ * F_ + (size_t)t * F_ + f;
#pragma unroll
  for (int o = 0; o < 64; ++o) {
    const size_t idx = base + (size_t)o * T_ * F_;
    out[idx] = (y[o] - mu) * inv * sgp[o] + sbt[o] + x[idx];
  }
}

// ---------------------------------------------------------------------------
extern "C" void kernel_launch(void* const* d_in, const int* in_sizes, int n_in,
                              void* d_out, int out_size, void* d_ws, size_t ws_size,
                              hipStream_t stream) {
  (void)in_sizes; (void)n_in; (void)out_size; (void)ws_size;
  const float* x    = (const float*)d_in[0];
  const float* Wq   = (const float*)d_in[1];
  const float* bq   = (const float*)d_in[2];
  const float* aq   = (const float*)d_in[3];
  const float* gq   = (const float*)d_in[4];
  const float* btq  = (const float*)d_in[5];
  const float* Wk   = (const float*)d_in[6];
  const float* bk   = (const float*)d_in[7];
  const float* ak   = (const float*)d_in[8];
  const float* gk   = (const float*)d_in[9];
  const float* btk  = (const float*)d_in[10];
  const float* Wv   = (const float*)d_in[11];
  const float* bv   = (const float*)d_in[12];
  const float* av   = (const float*)d_in[13];
  const float* gv   = (const float*)d_in[14];
  const float* btv  = (const float*)d_in[15];
  const float* Wp   = (const float*)d_in[16];
  const float* bp   = (const float*)d_in[17];
  const float* ap   = (const float*)d_in[18];
  const float* gp   = (const float*)d_in[19];
  const float* btp  = (const float*)d_in[20];

  // workspace partition (f16). Sizes (halfs):
  //   Qf,Kf: 8*3008*288 = 6,930,432 each ; Vt: 8*1040*3008 = 25,026,560 ;
  //   Of:    8*3000*1040 = 24,960,000   ; total ~128 MB.
  _Float16* Qf = (_Float16*)d_ws;
  _Float16* Kf = Qf + (size_t)B_ * H_ * TP_ * DP_;
  _Float16* Vt = Kf + (size_t)B_ * H_ * TP_ * DP_;
  _Float16* Of = Vt + (size_t)B_ * H_ * EF_ * TP_;

  pad_zero_kernel<<<(B_ * H_ * 12928) / 256, 256, 0, stream>>>(Qf, Kf, Vt);
  proj_kernel<<<(NPOS + 255) / 256, 256, 0, stream>>>(
      x, Wq, bq, aq, gq, btq, Wk, bk, ak, gk, btk, Wv, bv, av, gv, btv, Qf, Kf, Vt);
  attn_kernel<<<dim3(TP_ / 16, B_ * H_), 256, 0, stream>>>(Qf, Kf, Vt, Of);
  out_kernel<<<(NPOS + 255) / 256, 256, 0, stream>>>(x, Of, Wp, bp, ap, gp, btp,
                                                     (float*)d_out);
}
